// FeatureGNN_30150670418208
// MI455X (gfx1250) — matrix operand order repaired and checked by reading, compiled, not run on hardware
//
#include <hip/hip_runtime.h>
#include <cstring>
#include <cstdint>

// ---------------------------------------------------------------------------
// Types for CDNA5 WMMA (wave32): v_wmma_f32_16x16x32_bf16
// ---------------------------------------------------------------------------
typedef __bf16 bf16_t;
typedef __bf16 v16bf __attribute__((ext_vector_type(16)));
typedef float  v8f   __attribute__((ext_vector_type(8)));
typedef int    v4i   __attribute__((ext_vector_type(4)));

union FragBf { v16bf v; v4i q[2]; };

// Problem constants (match reference)
#define CN   50000
#define CE   200000
#define CB   512
#define CHID 512
#define CEIG 128
#define CKE  8

// ---------------------------------------------------------------------------
// Generic bf16 WMMA GEMM:  C[M,Nout] = act( A[M,K] @ Bt[Nout,K]^T + bias (+C) )
//   A may be row-gathered through rowIdx (edge-layer src/dst gathers).
//   Bt is the transposed weight, bf16, leading dim ldb.
//   Writes f32 (Cf) and/or bf16 (Cb) outputs.
// Block tile 128x128x32; 8 waves; wave tile 64x32 (4x2 WMMA 16x16 tiles).
// Staging uses CDNA5 async global->LDS DMA (ASYNCcnt) with double buffering:
// one s_wait_asynccnt + one workgroup barrier per K-step.
// REQUIREMENT: K % 8 == 0 (true for all GEMMs in this model).
// ---------------------------------------------------------------------------
#define TM 128
#define TN 128
#define TK 32
#define LDSK (TK + 8)   // pad: keeps 16B alignment (40 bf16 = 80B) & spreads banks

__device__ __forceinline__ void wait_async() {
#if __has_builtin(__builtin_amdgcn_s_wait_asynccnt)
    __builtin_amdgcn_s_wait_asynccnt(0);
#else
    asm volatile("s_wait_asynccnt 0x0" ::: "memory");
#endif
}

// Stage one 128x32 bf16 tile into LDS. Valid 16B segments go through the
// TDM-style async path (global_load_async_to_lds_b128, per-lane LDS/global
// addresses, no VGPR data); out-of-range segments are zero-filled with a
// direct 16B LDS store (DScnt, covered by __syncthreads).
__device__ __forceinline__ void stage_tile(bf16_t (* __restrict__ Ts)[LDSK],
                                           const bf16_t* __restrict__ G, int ld,
                                           const int* __restrict__ rowIdx,
                                           int rowBase, int k0,
                                           int rowLim, int Klim, int t)
{
    for (int s = t; s < TM * 4; s += 256) {
        const int r  = s >> 2;
        const int kk = (s & 3) << 3;
        const int gr = rowBase + r;
        const int gk = k0 + kk;
        if (gr < rowLim && gk < Klim) {        // Klim % 8 == 0 => full 16B valid
            long long ra = rowIdx ? (long long)rowIdx[gr] : (long long)gr;
            unsigned lds = (unsigned)(size_t)&Ts[r][kk];   // low 32b = LDS offset
            unsigned long long ga =
                (unsigned long long)(size_t)(G + (size_t)ra * ld + gk);
            asm volatile("global_load_async_to_lds_b128 %0, %1, off"
                         :: "v"(lds), "v"(ga) : "memory");
        } else {
            *(v4i*)&Ts[r][kk] = (v4i){0, 0, 0, 0};
        }
    }
}

__global__ __launch_bounds__(256)
void k_gemm(const bf16_t* __restrict__ A, int lda,
            const int* __restrict__ rowIdx,
            const bf16_t* __restrict__ Bt, int ldb,
            const float* __restrict__ bias,
            float* __restrict__ Cf,
            bf16_t* __restrict__ Cb, int ldc,
            int M, int Nout, int K,
            int accumulate, int act)
{
    __shared__ bf16_t As[2][TM][LDSK];
    __shared__ bf16_t Bs[2][TN][LDSK];

    const int t    = threadIdx.x;
    const int lane = t & 31;
    const int wave = t >> 5;
    const int row0 = blockIdx.y * TM;
    const int col0 = blockIdx.x * TN;

    const int wm = (wave >> 2) * 64;   // 0 or 64
    const int wn = (wave & 3)  * 32;   // 0,32,64,96

    v8f acc[4][2];
    const v8f vzero = {0.f,0.f,0.f,0.f,0.f,0.f,0.f,0.f};
#pragma unroll
    for (int i = 0; i < 4; i++)
#pragma unroll
        for (int j = 0; j < 2; j++) acc[i][j] = vzero;

    // ISA 7.12.2 fragment lane mapping
    const int mA  = lane & 15;            // A: M row within 16
    const int khA = (lane >> 4) << 3;     // A: K half offset (0 / 8)
    const int nB  = lane & 15;            // B: N col within 16
    const int kbB = (lane >> 4) << 3;     // B: K half offset (0 / 8)

    const int nk = (K + TK - 1) / TK;

    // prologue: stage tile 0 into buffer 0
    stage_tile(As[0], A,  lda, rowIdx,  row0, 0, M,    K, t);
    stage_tile(Bs[0], Bt, ldb, nullptr, col0, 0, Nout, K, t);

    for (int kt = 0; kt < nk; kt++) {
        wait_async();          // own async writes landed in LDS
        __syncthreads();       // everyone's writes visible; prev reads done
        const int cur = kt & 1;
        if (kt + 1 < nk) {     // overlap next tile's DMA with this tile's WMMAs
            stage_tile(As[cur ^ 1], A,  lda, rowIdx,  row0, (kt + 1) * TK, M,    K, t);
            stage_tile(Bs[cur ^ 1], Bt, ldb, nullptr, col0, (kt + 1) * TK, Nout, K, t);
        }
        bf16_t (*Ac)[LDSK] = As[cur];
        bf16_t (*Bc)[LDSK] = Bs[cur];

        FragBf fa[4], fb[2];
#pragma unroll
        for (int i = 0; i < 4; i++) {
            fa[i].q[0] = *(const v4i*)&Ac[wm + 16*i + mA][khA];
            fa[i].q[1] = *(const v4i*)&Ac[wm + 16*i + mA][16 + khA];
        }
#pragma unroll
        for (int j = 0; j < 2; j++) {
            fb[j].q[0] = *(const v4i*)&Bc[wn + 16*j + nB][kbB];
            fb[j].q[1] = *(const v4i*)&Bc[wn + 16*j + nB][16 + kbB];
        }
#pragma unroll
        for (int i = 0; i < 4; i++)
#pragma unroll
            for (int j = 0; j < 2; j++)
                acc[i][j] = __builtin_amdgcn_wmma_f32_16x16x32_bf16(
                    false, fa[i].v, false, fb[j].v, (short)0, acc[i][j],
                    false, false);
        // no trailing barrier: next iteration's top barrier protects reuse
    }

    // ---- epilogue: C/D layout (lane>=16 -> M+8), VGPR r -> M row ----
    const int rBase = (lane >> 4) << 3;
    const int cLane = lane & 15;
#pragma unroll
    for (int i = 0; i < 4; i++) {
#pragma unroll
        for (int j = 0; j < 2; j++) {
            int col = col0 + wn + 16*j + cLane;
            if (col >= Nout) continue;
            float bv = bias ? bias[col] : 0.f;
#pragma unroll
            for (int r = 0; r < 8; r++) {
                int row = row0 + wm + 16*i + rBase + r;
                if (row >= M) continue;
                size_t idx = (size_t)row * ldc + col;
                float v = acc[i][j][r] + bv;
                if (accumulate) v += Cf[idx];
                if (act) v = v * (1.f / (1.f + __expf(-v)));   // SiLU
                if (Cf) Cf[idx] = v;
                if (Cb) Cb[idx] = (bf16_t)v;
            }
        }
    }
}

// ---------------------------------------------------------------------------
// Elementwise / scatter helpers
// ---------------------------------------------------------------------------
__global__ void k_cast(const float* __restrict__ in, bf16_t* __restrict__ out, int n) {
    int i = blockIdx.x * 256 + threadIdx.x;
    if (i < n) out[i] = (bf16_t)in[i];
}

// W [Kd][Nd] f32 (row-major) -> out [Nd][Kd] bf16
__global__ void k_transpose(const float* __restrict__ W, bf16_t* __restrict__ out,
                            int Kd, int Nd) {
    int i = blockIdx.x * 256 + threadIdx.x;
    if (i >= Kd * Nd) return;
    int nrow = i / Kd;
    int k = i - nrow * Kd;
    out[i] = (bf16_t)W[(size_t)k * Nd + nrow];
}

__global__ void k_fill_u32(unsigned* p, unsigned v, int n) {
    int i = blockIdx.x * 256 + threadIdx.x;
    if (i < n) p[i] = v;
}

__global__ void k_counts(const int* __restrict__ batch, float* cnt, int n) {
    int i = blockIdx.x * 256 + threadIdx.x;
    if (i < n) atomicAdd(&cnt[batch[i]], 1.f);
}

__global__ void k_edge_meta(const int* __restrict__ src, const int* __restrict__ batch,
                            int* ebatch, float* ecnt, int e) {
    int i = blockIdx.x * 256 + threadIdx.x;
    if (i >= e) return;
    int b = batch[src[i]];
    ebatch[i] = b;
    atomicAdd(&ecnt[b], 1.f);
}

// SignNet phi0: rows = [sign(2)][node(N)][kk(8)], cols = 128
__global__ void k_phi0(const float* __restrict__ ev, const float* __restrict__ el,
                       const float* __restrict__ w0, const float* __restrict__ b0,
                       bf16_t* __restrict__ out, int Nn, int total) {
    int i = blockIdx.x * 256 + threadIdx.x;
    if (i >= total) return;
    int c = i & 127;
    int row = i >> 7;
    int sgn = row / (Nn * CKE);
    int rem = row - sgn * Nn * CKE;
    int node = rem >> 3;
    int kk = rem & 7;
    float v0 = ev[node * CKE + kk];
    if (sgn) v0 = -v0;
    float v1 = el[node * CKE + kk];
    out[i] = (bf16_t)(v0 * w0[c] + v1 * w0[128 + c] + b0[c]);
}

// sum over sign(2) x kk(8) -> [N,128]
__global__ void k_sign_sum(const bf16_t* __restrict__ in, bf16_t* __restrict__ out,
                           int Nn, int total) {
    int i = blockIdx.x * 256 + threadIdx.x;
    if (i >= total) return;
    int c = i & 127;
    int node = i >> 7;
    float s = 0.f;
#pragma unroll
    for (int sg = 0; sg < 2; sg++)
#pragma unroll
        for (int kk = 0; kk < 8; kk++)
            s += (float)in[(((size_t)(sg * Nn + node)) * CKE + kk) * CEIG + c];
    out[i] = (bf16_t)s;
}

// GINE message + scatter: agg[dst] += silu(hlin[src] + elin)
__global__ void k_msg(const float* __restrict__ hlin, const float* __restrict__ elin,
                      const int* __restrict__ src, const int* __restrict__ dst,
                      float* __restrict__ agg, int total) {
    int i = blockIdx.x * 256 + threadIdx.x;
    if (i >= total) return;
    int e = i >> 9;
    int c = i & 511;
    float v = hlin[(size_t)src[e] * CHID + c] + elin[i];
    v = v * (1.f / (1.f + __expf(-v)));
    atomicAdd(&agg[(size_t)dst[e] * CHID + c], v);
}

// pre = silu((1+eps)*hlin + agg) + oldh  (in-place into hlin) + seg sum/sumsq
__global__ void k_node_combine(float* __restrict__ hlin, const float* __restrict__ agg,
                               const float* __restrict__ oldh,
                               const float* __restrict__ epsArr, int layer,
                               const int* __restrict__ batch,
                               float* __restrict__ S1, float* __restrict__ S2, int total) {
    int i = blockIdx.x * 256 + threadIdx.x;
    if (i >= total) return;
    int node = i >> 9;
    int c = i & 511;
    float h = (1.f + epsArr[layer]) * hlin[i] + agg[i];
    h = h * (1.f / (1.f + __expf(-h)));
    float pre = h + oldh[i];
    hlin[i] = pre;
    int b = batch[node];
    atomicAdd(&S1[b * CHID + c], pre);
    atomicAdd(&S2[b * CHID + c], pre * pre);
}

// GraphNorm apply: var = S2/c + m^2(a^2-2a), out = gamma*(x-a*m)*rsqrt(var+eps)+beta
__global__ void k_node_apply(const float* __restrict__ pre,
                             const float* __restrict__ S1, const float* __restrict__ S2,
                             const float* __restrict__ cnt, const int* __restrict__ batch,
                             const float* __restrict__ gamma, const float* __restrict__ beta,
                             const float* __restrict__ alpha, const float* __restrict__ mask,
                             float* __restrict__ outf, bf16_t* __restrict__ outb, int total) {
    int i = blockIdx.x * 256 + threadIdx.x;
    if (i >= total) return;
    int node = i >> 9;
    int c = i & 511;
    int b = batch[node];
    float cc = fmaxf(cnt[b], 1.f);
    float m = S1[b * CHID + c] / cc;
    float a = alpha[c];
    float var = S2[b * CHID + c] / cc + m * m * (a * a - 2.f * a);
    float v = gamma[c] * (pre[i] - a * m) * rsqrtf(var + 1e-5f) + beta[c];
    v *= mask[node];
    outf[i] = v;
    outb[i] = (bf16_t)v;
}

__global__ void k_edge_combine(float* __restrict__ ez, const float* __restrict__ oldh,
                               const int* __restrict__ ebatch,
                               float* __restrict__ S1, float* __restrict__ S2, int total) {
    int i = blockIdx.x * 256 + threadIdx.x;
    if (i >= total) return;
    int e = i >> 9;
    int c = i & 511;
    float z = ez[i];
    z = z * (1.f / (1.f + __expf(-z)));
    float pre = z + oldh[i];
    ez[i] = pre;
    int b = ebatch[e];
    atomicAdd(&S1[b * CHID + c], pre);
    atomicAdd(&S2[b * CHID + c], pre * pre);
}

__global__ void k_edge_apply(const float* __restrict__ pre,
                             const float* __restrict__ S1, const float* __restrict__ S2,
                             const float* __restrict__ cnt, const int* __restrict__ ebatch,
                             const float* __restrict__ gamma, const float* __restrict__ beta,
                             const float* __restrict__ alpha,
                             float* __restrict__ outf, bf16_t* __restrict__ outb, int total) {
    int i = blockIdx.x * 256 + threadIdx.x;
    if (i >= total) return;
    int e = i >> 9;
    int c = i & 511;
    int b = ebatch[e];
    float cc = fmaxf(cnt[b], 1.f);
    float m = S1[b * CHID + c] / cc;
    float a = alpha[c];
    float var = S2[b * CHID + c] / cc + m * m * (a * a - 2.f * a);
    float v = gamma[c] * (pre[i] - a * m) * rsqrtf(var + 1e-5f) + beta[c];
    outf[i] = v;
    outb[i] = (bf16_t)v;
}

// order-preserving float<->uint for atomicMax-based segment max
__device__ inline unsigned d_f2ord(float f) {
    unsigned u = __float_as_uint(f);
    return (u & 0x80000000u) ? ~u : (u | 0x80000000u);
}
__device__ inline float d_ord2f(unsigned u) {
    return (u & 0x80000000u) ? __uint_as_float(u & 0x7fffffffu) : __uint_as_float(~u);
}

// one wave per node: gate = <node_h, gate_w> (+ mask penalty), segmax via atomicMax
__global__ void k_gate(const float* __restrict__ nh, const float* __restrict__ gw,
                       const float* __restrict__ mask, const int* __restrict__ batch,
                       float* __restrict__ gate, unsigned* __restrict__ segmax, int n) {
    int node = blockIdx.x * 8 + (threadIdx.x >> 5);
    int lane = threadIdx.x & 31;
    if (node >= n) return;
    float s = 0.f;
    for (int c = lane; c < CHID; c += 32) s += nh[(size_t)node * CHID + c] * gw[c];
#pragma unroll
    for (int off = 16; off > 0; off >>= 1) s += __shfl_down(s, off, 32);
    if (lane == 0) {
        if (!(mask[node] > 0.f)) s += -1e7f;
        gate[node] = s;
        atomicMax(&segmax[batch[node]], d_f2ord(s));
    }
}

__global__ void k_gatenorm(float* __restrict__ gate, const unsigned* __restrict__ segmax,
                           const int* __restrict__ batch, float* __restrict__ segsum, int n) {
    int i = blockIdx.x * 256 + threadIdx.x;
    if (i >= n) return;
    int b = batch[i];
    float g = __expf(gate[i] - d_ord2f(segmax[b]));
    gate[i] = g;
    atomicAdd(&segsum[b], g);
}

__global__ void k_wagg(const float* __restrict__ nh, const float* __restrict__ gate,
                       const float* __restrict__ segsum, const int* __restrict__ batch,
                       float* __restrict__ graph, int total) {
    int i = blockIdx.x * 256 + threadIdx.x;
    if (i >= total) return;
    int node = i >> 9;
    int c = i & 511;
    int b = batch[node];
    float w = gate[node] / fmaxf(segsum[b], 1e-20f);
    atomicAdd(&graph[(size_t)b * CHID + c], w * nh[i]);
}

__global__ void k_addcast(const float* __restrict__ a, const float* __restrict__ b,
                          bf16_t* __restrict__ out, int n) {
    int i = blockIdx.x * 256 + threadIdx.x;
    if (i < n) out[i] = (bf16_t)(a[i] + b[i]);
}

__global__ void k_final(const float* __restrict__ g, const float* __restrict__ r,
                        float* __restrict__ out, int n) {
    int i = blockIdx.x * 256 + threadIdx.x;
    if (i < n) out[i] = g[i] + r[i];
}

// ---------------------------------------------------------------------------
// Host orchestration
// ---------------------------------------------------------------------------
static inline unsigned h_f2ord(float f) {
    unsigned u;
    std::memcpy(&u, &f, 4);
    return (u & 0x80000000u) ? ~u : (u | 0x80000000u);
}

#define GS(n) dim3((((n) + 255) / 256)), dim3(256), 0, stream

extern "C" void kernel_launch(void* const* d_in, const int* in_sizes, int n_in,
                              void* d_out, int out_size, void* d_ws, size_t ws_size,
                              hipStream_t stream) {
    (void)in_sizes; (void)n_in; (void)out_size; (void)ws_size;
    const int N = CN, E = CE, Bn = CB, HID = CHID, EIG = CEIG;

    // Input order = setup_inputs() recursive dict insertion order.
    auto F = [&](int i) { return (const float*)d_in[i]; };
    const float *node_x = F(0), *edge_x = F(1), *covar = F(2);
    const float *eigv = F(3), *eigl = F(4), *maskp = F(5);
    const float *n_w1 = F(6),  *n_b1 = F(7),  *n_w2 = F(8),  *n_b2 = F(9);
    const float *e_w1 = F(10), *e_b1 = F(11), *e_w2 = F(12), *e_b2 = F(13);
    const float *c_w1 = F(14), *c_b1 = F(15), *c_w2 = F(16), *c_b2 = F(17);
    const float *phi0_w = F(18), *phi0_b = F(19), *phi_w = F(20), *phi_b = F(21);
    const float *rho_w = F(22), *rho_b = F(23), *rhof_w = F(24), *rhof_b = F(25);
    const float *nl_lin1_w = F(26), *nl_lin1_b = F(27), *nl_elin_w = F(28), *nl_elin_b = F(29);
    const float *nl_eps = F(30), *nl_gamma = F(31), *nl_beta = F(32), *nl_alpha = F(33);
    const float *el_w = F(34), *el_b = F(35), *el_gamma = F(36), *el_beta = F(37), *el_alpha = F(38);
    const float *gate_w = F(39), *covres_w = F(40), *covres_b = F(41);
    const int* batch = (const int*)d_in[42];
    const int* eidx  = (const int*)d_in[43];
    const int* esrc = eidx;
    const int* edst = eidx + E;

    // ---- workspace bump allocator ----
    char* base = (char*)d_ws;
    size_t off = 0;
    auto alloc = [&](size_t bytes) -> void* {
        void* p = base + off;
        off = (off + bytes + 255) & ~(size_t)255;
        return p;
    };
    const size_t NH = (size_t)N * HID, EH = (size_t)E * HID, BH = (size_t)Bn * HID;
    const size_t M0 = (size_t)2 * N * CKE;   // signnet rows (both signs)

    // bf16 activations
    bf16_t* bx_node = (bf16_t*)alloc((size_t)N * 64 * 2);
    bf16_t* bx_edge = (bf16_t*)alloc((size_t)E * 16 * 2);
    bf16_t* bx_cov  = (bf16_t*)alloc((size_t)Bn * 32 * 2);
    bf16_t* sgnA = (bf16_t*)alloc(M0 * EIG * 2);
    bf16_t* sgnB = (bf16_t*)alloc(M0 * EIG * 2);
    bf16_t* rhoA = (bf16_t*)alloc((size_t)N * EIG * 2);
    bf16_t* rhoB = (bf16_t*)alloc((size_t)N * EIG * 2);
    bf16_t* ntmp_bf = (bf16_t*)alloc(NH * 2);
    bf16_t* nh_bf   = (bf16_t*)alloc(NH * 2);
    bf16_t* etmp_bf = (bf16_t*)alloc(EH * 2);
    bf16_t* eh_bf   = (bf16_t*)alloc(EH * 2);
    bf16_t* gcov_bf = (bf16_t*)alloc(BH * 2);
    bf16_t* covtmp_bf = (bf16_t*)alloc(BH * 2);
    // f32 buffers
    float* nh_f  = (float*)alloc(NH * 4);
    float* eh_f  = (float*)alloc(EH * 4);
    float* hlin  = (float*)alloc(NH * 4);   // reused in-place as node pre-norm
    float* elin  = (float*)alloc(EH * 4);
    float* agg   = (float*)alloc(NH * 4);
    float* ez    = (float*)alloc(EH * 4);   // reused in-place as edge pre-norm
    float* S1    = (float*)alloc(BH * 4);
    float* S2    = (float*)alloc(BH * 4);
    float* ncnt  = (float*)alloc(Bn * 4);
    float* ecnt  = (float*)alloc(Bn * 4);
    float* covh  = (float*)alloc(BH * 4);
    float* gate  = (float*)alloc((size_t)N * 4);
    float* segsum = (float*)alloc(Bn * 4);
    float* graph = (float*)alloc(BH * 4);
    float* res   = (float*)alloc(BH * 4);
    int* ebatch  = (int*)alloc((size_t)E * 4);
    unsigned* segmax = (unsigned*)alloc(Bn * 4);
    // transposed bf16 weights [Nout][K]
    bf16_t* wt_node1 = (bf16_t*)alloc((size_t)512 * 64 * 2);
    bf16_t* wt_node2 = (bf16_t*)alloc((size_t)512 * 512 * 2);
    bf16_t* wt_edge1 = (bf16_t*)alloc((size_t)512 * 16 * 2);
    bf16_t* wt_edge2 = (bf16_t*)alloc((size_t)512 * 512 * 2);
    bf16_t* wt_cov1  = (bf16_t*)alloc((size_t)512 * 32 * 2);
    bf16_t* wt_cov2  = (bf16_t*)alloc((size_t)512 * 512 * 2);
    bf16_t* wt_phi   = (bf16_t*)alloc((size_t)2 * 128 * 128 * 2);
    bf16_t* wt_rho   = (bf16_t*)alloc((size_t)2 * 128 * 128 * 2);
    bf16_t* wt_rhof  = (bf16_t*)alloc((size_t)512 * 128 * 2);
    bf16_t* wt_lin1  = (bf16_t*)alloc((size_t)6 * 512 * 512 * 2);
    bf16_t* wt_elin  = (bf16_t*)alloc((size_t)6 * 512 * 512 * 2);
    bf16_t* wt_el    = (bf16_t*)alloc((size_t)5 * 512 * 1536 * 2);
    bf16_t* wt_covres = (bf16_t*)alloc((size_t)512 * 512 * 2);

    auto transpose = [&](const float* W, int Kd, int Nd, bf16_t* out) {
        int n = Kd * Nd;
        k_transpose<<<GS(n)>>>(W, out, Kd, Nd);
    };
    auto gemm = [&](const bf16_t* A, int lda, const int* rix,
                    const bf16_t* Bt, int ldb, const float* bias,
                    float* Cf, bf16_t* Cb, int ldc,
                    int M, int Nd, int K, int acc, int act) {
        dim3 g((Nd + TN - 1) / TN, (M + TM - 1) / TM);
        k_gemm<<<g, 256, 0, stream>>>(A, lda, rix, Bt, ldb, bias, Cf, Cb, ldc,
                                      M, Nd, K, acc, act);
    };

    // ---- 0. weight prep (bf16 + transpose) & input casts ----
    transpose(n_w1, 64, 512, wt_node1);
    transpose(n_w2, 512, 512, wt_node2);
    transpose(e_w1, 16, 512, wt_edge1);
    transpose(e_w2, 512, 512, wt_edge2);
    transpose(c_w1, 32, 512, wt_cov1);
    transpose(c_w2, 512, 512, wt_cov2);
    for (int d = 0; d < 2; d++) {
        transpose(phi_w + (size_t)d * 128 * 128, 128, 128, wt_phi + (size_t)d * 128 * 128);
        transpose(rho_w + (size_t)d * 128 * 128, 128, 128, wt_rho + (size_t)d * 128 * 128);
    }
    transpose(rhof_w, 128, 512, wt_rhof);
    for (int i = 0; i < 6; i++) {
        transpose(nl_lin1_w + (size_t)i * 512 * 512, 512, 512, wt_lin1 + (size_t)i * 512 * 512);
        transpose(nl_elin_w + (size_t)i * 512 * 512, 512, 512, wt_elin + (size_t)i * 512 * 512);
    }
    for (int i = 0; i < 5; i++)
        transpose(el_w + (size_t)i * 1536 * 512, 1536, 512, wt_el + (size_t)i * 512 * 1536);
    transpose(covres_w, 512, 512, wt_covres);
    k_cast<<<GS(N * 64)>>>(node_x, bx_node, N * 64);
    k_cast<<<GS(E * 16)>>>(edge_x, bx_edge, E * 16);
    k_cast<<<GS(Bn * 32)>>>(covar, bx_cov, Bn * 32);

    // ---- counts / edge metadata ----
    hipMemsetAsync(ncnt, 0, Bn * 4, stream);
    hipMemsetAsync(ecnt, 0, Bn * 4, stream);
    k_counts<<<GS(N)>>>(batch, ncnt, N);
    k_edge_meta<<<GS(E)>>>(esrc, batch, ebatch, ecnt, E);

    // ---- embeddings (WMMA GEMMs) ----
    gemm(bx_node, 64, nullptr, wt_node1, 64, n_b1, nullptr, ntmp_bf, 512, N, 512, 64, 0, 1);
    gemm(ntmp_bf, 512, nullptr, wt_node2, 512, n_b2, nh_f, nullptr, 512, N, 512, 512, 0, 0);
    gemm(bx_edge, 16, nullptr, wt_edge1, 16, e_b1, nullptr, etmp_bf, 512, E, 512, 16, 0, 1);
    gemm(etmp_bf, 512, nullptr, wt_edge2, 512, e_b2, eh_f, eh_bf, 512, E, 512, 512, 0, 0);
    gemm(bx_cov, 32, nullptr, wt_cov1, 32, c_b1, nullptr, covtmp_bf, 512, Bn, 512, 32, 0, 1);
    gemm(covtmp_bf, 512, nullptr, wt_cov2, 512, c_b2, covh, nullptr, 512, Bn, 512, 512, 0, 0);

    // ---- SignNet ----
    k_phi0<<<GS((int)(M0 * EIG))>>>(eigv, eigl, phi0_w, phi0_b, sgnA, N, (int)(M0 * EIG));
    gemm(sgnA, 128, nullptr, wt_phi, 128, phi_b, nullptr, sgnB, 128, (int)M0, 128, 128, 0, 1);
    gemm(sgnB, 128, nullptr, wt_phi + 128 * 128, 128, phi_b + 128, nullptr, sgnA, 128, (int)M0, 128, 128, 0, 1);
    k_sign_sum<<<GS(N * EIG)>>>(sgnA, rhoA, N, N * EIG);
    gemm(rhoA, 128, nullptr, wt_rho, 128, rho_b, nullptr, rhoB, 128, N, 128, 128, 0, 1);
    gemm(rhoB, 128, nullptr, wt_rho + 128 * 128, 128, rho_b + 128, nullptr, rhoA, 128, N, 128, 128, 0, 1);
    // node_h += signnet (accumulate) and emit bf16 copy
    gemm(rhoA, 128, nullptr, wt_rhof, 128, rhof_b, nh_f, nh_bf, 512, N, 512, 128, 1, 0);

    // ---- GNN layers ----
    auto node_layer = [&](int i) {
        gemm(nh_bf, 512, nullptr, wt_lin1 + (size_t)i * 512 * 512, 512,
             nl_lin1_b + i * 512, hlin, nullptr, 512, N, 512, 512, 0, 0);
        gemm(eh_bf, 512, nullptr, wt_elin + (size_t)i * 512 * 512, 512,
             nl_elin_b + i * 512, elin, nullptr, 512, E, 512, 512, 0, 0);
        hipMemsetAsync(agg, 0, NH * 4, stream);
        k_msg<<<GS((int)EH)>>>(hlin, elin, esrc, edst, agg, (int)EH);
        hipMemsetAsync(S1, 0, BH * 4, stream);
        hipMemsetAsync(S2, 0, BH * 4, stream);
        k_node_combine<<<GS((int)NH)>>>(hlin, agg, nh_f, nl_eps, i, batch, S1, S2, (int)NH);
        k_node_apply<<<GS((int)NH)>>>(hlin, S1, S2, ncnt, batch,
                                      nl_gamma + i * 512, nl_beta + i * 512,
                                      nl_alpha + i * 512, maskp, nh_f, nh_bf, (int)NH);
    };
    auto edge_layer = [&](int i) {
        const bf16_t* W = wt_el + (size_t)i * 512 * 1536;   // [512][1536]: K-blocks e|src|dst
        gemm(eh_bf, 512, nullptr, W, 1536, el_b + i * 512, ez, nullptr, 512, E, 512, 512, 0, 0);
        gemm(nh_bf, 512, esrc, W + 512, 1536, nullptr, ez, nullptr, 512, E, 512, 512, 1, 0);
        gemm(nh_bf, 512, edst, W + 1024, 1536, nullptr, ez, nullptr, 512, E, 512, 512, 1, 0);
        hipMemsetAsync(S1, 0, BH * 4, stream);
        hipMemsetAsync(S2, 0, BH * 4, stream);
        k_edge_combine<<<GS((int)EH)>>>(ez, eh_f, ebatch, S1, S2, (int)EH);
        k_edge_apply<<<GS((int)EH)>>>(ez, S1, S2, ecnt, ebatch,
                                      el_gamma + i * 512, el_beta + i * 512,
                                      el_alpha + i * 512, eh_f, eh_bf, (int)EH);
    };
    for (int i = 0; i < 5; i++) {
        node_layer(i);
        edge_layer(i);
    }
    node_layer(5);

    // ---- attentional aggregation (segment softmax) ----
    k_fill_u32<<<GS(Bn)>>>(segmax, h_f2ord(-3e38f), Bn);
    hipMemsetAsync(segsum, 0, Bn * 4, stream);
    hipMemsetAsync(graph, 0, BH * 4, stream);
    k_gate<<<dim3((N + 7) / 8), dim3(256), 0, stream>>>(nh_f, gate_w, maskp, batch,
                                                        gate, segmax, N);
    k_gatenorm<<<GS(N)>>>(gate, segmax, batch, segsum, N);
    k_wagg<<<GS((int)NH)>>>(nh_f, gate, segsum, batch, graph, (int)NH);

    // ---- covariate residual block ----
    k_addcast<<<GS((int)BH)>>>(graph, covh, gcov_bf, (int)BH);
    gemm(gcov_bf, 512, nullptr, wt_covres, 512, covres_b, res, nullptr, 512, Bn, 512, 512, 0, 1);
    k_final<<<GS((int)BH)>>>(graph, res, (float*)d_out, (int)BH);
}
